// CausalSelfAttention_59923383713872
// MI455X (gfx1250) — compile-verified, matrix operand order checked by reading, not run on hardware
//
#include <hip/hip_runtime.h>
#include <hip/hip_bf16.h>

// CDNA5 / gfx1250, wave32. bf16 WMMA (V_WMMA_F32_16X16X32_BF16) everywhere.
// Shapes: B=4, T=2048, C=1024, H=16, D=64. M = B*T = 8192.
// Workspace layout (d_ws):
//   [0, 50331648)              : qkv  bf16 [8192, 3072]
//   [50331648, +16MiB)         : attn bf16 [8192, 1024]

typedef __attribute__((ext_vector_type(16))) __bf16 v16bf;
typedef __attribute__((ext_vector_type(8)))  float  v8f;
typedef int v4i __attribute__((vector_size(16)));    // matches async-LDS builtin pointee

#define ASG __attribute__((address_space(1)))
#define ASL __attribute__((address_space(3)))

#if defined(__has_builtin)
#if __has_builtin(__builtin_amdgcn_global_load_async_to_lds_b128) && \
    __has_builtin(__builtin_amdgcn_s_wait_asynccnt)
#define USE_ASYNC_LDS 1
#endif
#endif

__device__ __forceinline__ __bf16 to_bf16(float v)  { return (__bf16)v; }
__device__ __forceinline__ __bf16 to_bf16(__bf16 v) { return v; }

// ---------------------------------------------------------------------------
// Tiled GEMM: Y[M,N] = X[M,K] @ W[K,N], bf16 WMMA, f32 accumulate.
// Block 256 threads = 8 waves. Tile 128x128, K-step 32.
// W tile stored TRANSPOSED in LDS so B-fragment gathers are K-contiguous
// (2x ds_load_b128 per fragment instead of 16x ds_load_u16).
// ---------------------------------------------------------------------------
template <typename XT, typename OutT>
__global__ __launch_bounds__(256) void gemm_wmma(
    const XT* __restrict__ X, const float* __restrict__ W, OutT* __restrict__ Y,
    int M, int N, int K)
{
    __shared__ __bf16 As [128][32 + 8];  // [m][k]
    __shared__ __bf16 BsT[128][32 + 8];  // [n][k]  (transposed W tile)

    const int tid   = threadIdx.x;
    const int lane  = tid & 31;
    const int wave  = tid >> 5;          // 0..7
    const int waveM = wave & 3;          // 32-row stripe
    const int waveN = wave >> 2;         // 64-col stripe
    const int am    = lane & 15;         // A: M within tile / B: N within tile
    const int akb   = (lane >> 4) << 3;  // K sub-base: 0 or 8 (ISA 16-bit A layout)
    const int rowHf = (lane >> 4) << 3;  // C/D: row = r + rowHf

    const int m0 = blockIdx.y * 128;
    const int n0 = blockIdx.x * 128;

    // Cooperative loaders: A 128x32 (16 elems/thread), W 32x128 (16 elems/thread)
    const int ar = tid >> 1;             // 0..127
    const int ac = (tid & 1) * 16;
    const int br = tid >> 3;             // 0..31 (k)
    const int bc = (tid & 7) * 16;       // n within 128

    v8f acc[2][4] = {};

    for (int k0 = 0; k0 < K; k0 += 32) {
        __syncthreads();
        #pragma unroll
        for (int c = 0; c < 16; ++c)
            As[ar][ac + c] = to_bf16(X[(long)(m0 + ar) * K + (k0 + ac + c)]);
        #pragma unroll
        for (int c = 0; c < 16; ++c)     // transpose on store: [n][k]
            BsT[bc + c][br] = to_bf16((float)W[(long)(k0 + br) * N + (n0 + bc + c)]);
        if (k0 + 32 < K) {               // prefetch next K-step (global_prefetch_b8)
            __builtin_prefetch(&X[(long)(m0 + ar) * K + (k0 + 32 + ac)], 0, 1);
            __builtin_prefetch(&W[(long)(k0 + 32 + br) * N + (n0 + bc)], 0, 1);
        }
        __syncthreads();

        // A fragments (16x32, ISA A-layout: elem e -> K = akb + e + (e<8?0:8))
        v16bf af[2];
        #pragma unroll
        for (int i = 0; i < 2; ++i)
            #pragma unroll
            for (int e = 0; e < 16; ++e) {
                int kk = akb + e + ((e < 8) ? 0 : 8);
                af[i][e] = As[waveM * 32 + i * 16 + am][kk];
            }

        #pragma unroll
        for (int j = 0; j < 4; ++j) {
            v16bf bf;
            #pragma unroll
            for (int e = 0; e < 16; ++e) {
                int kk = akb + e + ((e < 8) ? 0 : 8);
                bf[e] = BsT[waveN * 64 + j * 16 + am][kk];   // K-contiguous
            }
            #pragma unroll
            for (int i = 0; i < 2; ++i)
                acc[i][j] = __builtin_amdgcn_wmma_f32_16x16x32_bf16(
                    false, af[i], false, bf, (short)0, acc[i][j], false, false);
        }
    }

    // C/D layout: VGPR r -> row r + rowHf, col = lane&15
    #pragma unroll
    for (int i = 0; i < 2; ++i)
        #pragma unroll
        for (int j = 0; j < 4; ++j)
            #pragma unroll
            for (int r = 0; r < 8; ++r) {
                int gm = m0 + waveM * 32 + i * 16 + r + rowHf;
                int gn = n0 + waveN * 64 + j * 16 + am;
                Y[(long)gm * N + gn] = (OutT)acc[i][j][r];
            }
}

// ---------------------------------------------------------------------------
// Flash-style causal attention over bf16 qkv.
// Grid: (T/64, B*H). Block: 128 threads = 4 waves; wave w owns query rows
// qbase + w*16 .. +15.
//  - K tile (key-major) staged via GLOBAL_LOAD_ASYNC_TO_LDS_B128 (ASYNCcnt).
//  - V tile staged TRANSPOSED ([dim][key]) so P@V B-fragments are K-contiguous.
//  - Softmax in exp2 domain; 0.125*log2(e) folded into bf16 Q once.
// ---------------------------------------------------------------------------
__global__ __launch_bounds__(128) void attn_kernel(
    const __bf16* __restrict__ qkv, __bf16* __restrict__ attn_out)
{
    const int T = 2048, C = 1024;
    const int tid   = threadIdx.x;
    const int lane  = tid & 31;
    const int wave  = tid >> 5;            // 0..3
    const int qtile = blockIdx.x;          // 64 query rows per block
    const int b     = blockIdx.y >> 4;
    const int h     = blockIdx.y & 15;
    const int qbase = qtile * 64;
    const int am    = lane & 15;
    const int akb   = (lane >> 4) << 3;
    const int rowHf = (lane >> 4) << 3;

    __shared__ __bf16 Ks [64][64 + 8];     // [key][dim]
    __shared__ __bf16 VsT[64][64 + 8];     // [dim][key]  (transposed V tile)
    __shared__ __bf16 Ps[4][16][64 + 8];   // per-wave P staging (C-layout -> A-layout)

    const long rs      = 3 * C;            // qkv row stride
    const long base_bt = (long)b * T;

    // Q fragments (A-layout), loaded once, pre-scaled by log2(e)/sqrt(D)
    const float qscale = 0.125f * 1.442695041f;
    const int qrow = qbase + wave * 16 + am;
    v16bf qf[2];
    #pragma unroll
    for (int kc = 0; kc < 2; ++kc)
        #pragma unroll
        for (int e = 0; e < 16; ++e) {
            int kk = akb + e + ((e < 8) ? 0 : 8);
            float q = (float)qkv[(base_bt + qrow) * rs + h * 64 + kc * 32 + kk];
            qf[kc][e] = (__bf16)(q * qscale);
        }

    v8f o[4] = {};
    float mrow[8], lrow[8];
    #pragma unroll
    for (int r = 0; r < 8; ++r) { mrow[r] = -3.0e38f; lrow[r] = 0.f; }

    // staging coordinates: thread -> row tid/2, 32-col half
    const int ldr  = tid >> 1;
    const int ldc0 = (tid & 1) * 32;

    for (int kt = 0; kt <= qtile; ++kt) {
        __syncthreads();                   // previous tile fully consumed
        {
            long krow = (base_bt + kt * 64 + ldr) * rs + C     + h * 64 + ldc0;
            long vrow = (base_bt + kt * 64 + ldr) * rs + 2 * C + h * 64 + ldc0;
#ifdef USE_ASYNC_LDS
            #pragma unroll
            for (int c8 = 0; c8 < 4; ++c8)
                __builtin_amdgcn_global_load_async_to_lds_b128(
                    (ASG v4i*)(qkv + krow + c8 * 8),
                    (ASL v4i*)&Ks[ldr][ldc0 + c8 * 8], 0, 0);
#else
            #pragma unroll
            for (int c = 0; c < 32; ++c)
                Ks[ldr][ldc0 + c] = qkv[krow + c];
#endif
            // V: contiguous global read, transposed LDS store [dim][key]
            #pragma unroll
            for (int c = 0; c < 32; ++c)
                VsT[ldc0 + c][ldr] = qkv[vrow + c];
#ifdef USE_ASYNC_LDS
            __builtin_amdgcn_s_wait_asynccnt(0);
#endif
        }
        __syncthreads();

        // S = (Q*scale) K^T  (B-frag: B[kk][n] = Ks[key = tj*16+n][d = kc*32+kk])
        v8f s[4] = {};
        #pragma unroll
        for (int tj = 0; tj < 4; ++tj)
            #pragma unroll
            for (int kc = 0; kc < 2; ++kc) {
                v16bf bf;
                #pragma unroll
                for (int e = 0; e < 16; ++e) {
                    int kk = akb + e + ((e < 8) ? 0 : 8);
                    bf[e] = Ks[tj * 16 + am][kc * 32 + kk];
                }
                s[tj] = __builtin_amdgcn_wmma_f32_16x16x32_bf16(
                    false, qf[kc], false, bf, (short)0, s[tj], false, false);
            }

        // causal mask: only the diagonal tile needs it
        if (kt == qtile) {
            #pragma unroll
            for (int tj = 0; tj < 4; ++tj)
                #pragma unroll
                for (int r = 0; r < 8; ++r) {
                    int qi = wave * 16 + r + rowHf;   // local query in [0,64)
                    int kj = tj * 16 + am;            // local key   in [0,64)
                    if (kj > qi) s[tj][r] = -3.0e38f;
                }
        }

        // online softmax in exp2 domain (rows across 16 lanes; xor-reduce)
        float alpha[8];
        #pragma unroll
        for (int r = 0; r < 8; ++r) {
            float mx = fmaxf(fmaxf(s[0][r], s[1][r]), fmaxf(s[2][r], s[3][r]));
            #pragma unroll
            for (int off = 1; off < 16; off <<= 1)
                mx = fmaxf(mx, __shfl_xor(mx, off, 32));
            float mnew = fmaxf(mrow[r], mx);
            alpha[r] = __builtin_amdgcn_exp2f(mrow[r] - mnew);
            mrow[r] = mnew;
            float rsum = 0.f;
            #pragma unroll
            for (int tj = 0; tj < 4; ++tj) {
                float p = __builtin_amdgcn_exp2f(s[tj][r] - mnew);
                s[tj][r] = p;
                rsum += p;
            }
            #pragma unroll
            for (int off = 1; off < 16; off <<= 1)
                rsum += __shfl_xor(rsum, off, 32);
            lrow[r] = lrow[r] * alpha[r] + rsum;
        }

        // C-layout P -> LDS -> A-layout fragments (wave-private region; LDS is
        // in-order within a wave, no barrier needed)
        #pragma unroll
        for (int tj = 0; tj < 4; ++tj)
            #pragma unroll
            for (int r = 0; r < 8; ++r)
                Ps[wave][r + rowHf][tj * 16 + am] = (__bf16)s[tj][r];

        v16bf pf[2];
        #pragma unroll
        for (int kc = 0; kc < 2; ++kc)
            #pragma unroll
            for (int e = 0; e < 16; ++e) {
                int kk = akb + e + ((e < 8) ? 0 : 8);
                pf[kc][e] = Ps[wave][am][kc * 32 + kk];
            }

        // rescale O, then O += P @ V  (B-frag from VsT: K-contiguous)
        #pragma unroll
        for (int td = 0; td < 4; ++td)
            #pragma unroll
            for (int r = 0; r < 8; ++r)
                o[td][r] *= alpha[r];

        #pragma unroll
        for (int td = 0; td < 4; ++td)
            #pragma unroll
            for (int kc = 0; kc < 2; ++kc) {
                v16bf vf;
                #pragma unroll
                for (int e = 0; e < 16; ++e) {
                    int kk = akb + e + ((e < 8) ? 0 : 8);
                    vf[e] = VsT[td * 16 + am][kc * 32 + kk];
                }
                o[td] = __builtin_amdgcn_wmma_f32_16x16x32_bf16(
                    false, pf[kc], false, vf, (short)0, o[td], false, false);
            }
    }

    // normalize and store merged-head output [B*T, C] in bf16
    #pragma unroll
    for (int td = 0; td < 4; ++td)
        #pragma unroll
        for (int r = 0; r < 8; ++r) {
            int qi = qbase + wave * 16 + r + rowHf;
            float val = o[td][r] / lrow[r];
            attn_out[(base_bt + qi) * C + h * 64 + td * 16 + am] = (__bf16)val;
        }
}

// ---------------------------------------------------------------------------
extern "C" void kernel_launch(void* const* d_in, const int* in_sizes, int n_in,
                              void* d_out, int out_size, void* d_ws, size_t ws_size,
                              hipStream_t stream) {
    const int Bv = 4, Tv = 2048, Cv = 1024, Hv = 16;
    const int M = Bv * Tv;                 // 8192

    const float* x      = (const float*)d_in[0];   // [B,T,C]
    const float* w_qkv  = (const float*)d_in[1];   // [C,3C]
    const float* w_proj = (const float*)d_in[2];   // [C,C]
    float* out          = (float*)d_out;           // [B,T,C]

    __bf16* qkv  = (__bf16*)d_ws;                                    // [M,3C] bf16
    __bf16* attn = (__bf16*)((char*)d_ws + (size_t)M * 3 * Cv * 2);  // [M,C]  bf16

    // 1) qkv = x @ w_qkv    (f32 in -> bf16 out)
    dim3 g1(3 * Cv / 128, M / 128);
    gemm_wmma<float, __bf16><<<g1, 256, 0, stream>>>(x, w_qkv, qkv, M, 3 * Cv, Cv);

    // 2) causal flash attention over qkv -> attn (bf16)
    dim3 g2(Tv / 64, Bv * Hv);
    attn_kernel<<<g2, 128, 0, stream>>>(qkv, attn);

    // 3) out = attn @ w_proj (bf16 in -> f32 out)
    dim3 g3(Cv / 128, M / 128);
    gemm_wmma<__bf16, float><<<g3, 256, 0, stream>>>(attn, w_proj, out, M, Cv, Cv);
}
